// HoloDistillLoss_721554506547
// MI455X (gfx1250) — matrix operand order repaired and checked by reading, not compiled, not run on hardware
//
#include <hip/hip_runtime.h>
#include <hip/hip_bf16.h>
#include <math.h>
#include <cstddef>
#include <cstdint>

typedef _Float16 half_t;
typedef __attribute__((ext_vector_type(16))) _Float16 v16h;
typedef __attribute__((ext_vector_type(8)))  float    v8f;

#define TEMPERATURE 0.07f
#define OT_EPS      0.1f
#define OT_ITERS    20

#define BB    64
#define NN    256
#define HS    768
#define HT    1024
#define DREP  768
#define MROWS (BB*NN)   /* 16384 */

// accumulator slots (f32) at start of workspace
#define ACC_CONTRAST 0
#define ACC_DS_SUM   1
#define ACC_DT_SUM   2
#define ACC_DIST     3
#define ACC_ANGLE    4
#define ACC_FEAT     5
#define ACC_STRUCT   6

// GEMM epilogue selectors
#define EPI_1M_F32   0   // OutF = 1 - c
#define EPI_1M_F16   1   // OutH = 1 - c
#define EPI_F16      2   // OutH = c
#define EPI_BIAS_F16 3   // OutH = c + bias[n]
#define EPI_STRUCT   4   // fused (Cs - c)^2 * mu[m]*mu[n] reduction

// ---------------------------------------------------------------- helpers

__device__ inline float huber_f(float d) {
  float a = fabsf(d);
  return a < 1.0f ? 0.5f * a * a : a - 0.5f;
}

template <int TPB>
__device__ inline float block_reduce(float v) {
  __shared__ float buf[TPB];
  int t = threadIdx.x;
  buf[t] = v; __syncthreads();
#pragma unroll
  for (int s = TPB >> 1; s > 0; s >>= 1) {
    if (t < s) buf[t] += buf[t + s];
    __syncthreads();
  }
  float r = buf[0]; __syncthreads();
  return r;
}

// --- CDNA5 async copy global->LDS (ASYNCcnt) ----------------------------
// LDS aperture flat address: addr[31:0] is the LDS byte offset (ISA 10.2).
__device__ inline unsigned lds_off_u32(const void* p) {
  return (unsigned)(uintptr_t)p;
}
__device__ inline void async_copy_b128(unsigned lds_byte_off, const void* gaddr) {
  asm volatile("global_load_async_to_lds_b128 %0, %1, off"
               :: "v"(lds_byte_off), "v"(gaddr) : "memory");
}
template <int N>
__device__ inline void wait_asynccnt() {
  asm volatile("s_wait_asynccnt %0" :: "n"(N));
}

// --- WMMA fragment loaders, contiguous k order, same in A and B (wave32) --------
// A fragment: A row-major [M x K], tile at (row0,k0), leading dim lda.
// lane L: m = L&15, khalf = L>>4;  k = h + 16*khalf
__device__ inline v16h load_a_frag(const half_t* A, int lda, int row0, int k0, int lane) {
  const half_t* p = A + (size_t)(row0 + (lane & 15)) * lda + k0 + ((lane >> 4) << 4);
  v16h a;
#pragma unroll
  for (int h = 0; h < 16; ++h) a[h] = p[h];
  return a;
}

// B fragment (transposed): B[k][n] = X[n][k], X row-major [N x K] -> contiguous in k
__device__ inline v16h load_b_frag_t(const half_t* X, int ldx, int k0, int n0, int lane) {
  const half_t* p = X + (size_t)(n0 + (lane & 15)) * ldx + k0 + ((lane >> 4) << 4);
  v16h b;
#pragma unroll
  for (int h = 0; h < 16; ++h) b[h] = p[h];
  return b;
}

#define WMMA_F16(a, b, c) \
  __builtin_amdgcn_wmma_f32_16x16x32_f16(false, (a), false, (b), (short)0, (c), false, false)

// ---------------------------------------------------------------- small kernels

__global__ void zero_acc_kernel(float* acc) {
  if (threadIdx.x < 32) acc[threadIdx.x] = 0.0f;
}

// transpose proj_w (HS x HT, f32) -> Wt (HT x HS, f16)
__global__ void transpose_w_kernel(const float* W, half_t* Wt) {
  __shared__ float tile[32][33];
  int bx = blockIdx.x, by = blockIdx.y;        // bx: HT/32, by: HS/32
  int tx = threadIdx.x & 31, ty = threadIdx.x >> 5;   // 32x8
  for (int r = ty; r < 32; r += 8)
    tile[r][tx] = W[(size_t)(by * 32 + r) * HT + bx * 32 + tx];
  __syncthreads();
  for (int r = ty; r < 32; r += 8)
    Wt[(size_t)(bx * 32 + r) * HS + by * 32 + tx] = (half_t)tile[tx][r];
}

// l2-normalize the first `dim` columns of each row (row stride in_stride)
__global__ void norm_slice_f32(const float* in, int in_stride, int dim,
                               float* out, int out_stride) {
  int row = blockIdx.x;
  const float* src = in + (size_t)row * in_stride;
  float ss = 0.0f;
  for (int d = threadIdx.x; d < dim; d += blockDim.x) { float x = src[d]; ss += x * x; }
  ss = block_reduce<256>(ss);
  float inv = 1.0f / fmaxf(sqrtf(ss), 1e-12f);
  float* dst = out + (size_t)row * out_stride;
  for (int d = threadIdx.x; d < dim; d += blockDim.x) dst[d] = src[d] * inv;
}

// row l2norm -> f16 (out_norm); optional raw f16 copy (out_raw)
__global__ void rownorm_f32_to_f16(const float* in, int H, half_t* out_norm, half_t* out_raw) {
  int row = blockIdx.x;
  const float* src = in + (size_t)row * H;
  float ss = 0.0f;
  for (int d = threadIdx.x; d < H; d += blockDim.x) { float x = src[d]; ss += x * x; }
  ss = block_reduce<256>(ss);
  float inv = 1.0f / fmaxf(sqrtf(ss), 1e-12f);
  half_t* dn = out_norm + (size_t)row * H;
  for (int d = threadIdx.x; d < H; d += blockDim.x) {
    float x = src[d];
    dn[d] = (half_t)(x * inv);
    if (out_raw) out_raw[(size_t)row * H + d] = (half_t)x;
  }
}

__global__ void rownorm_inplace_f16(half_t* data, int H) {
  int row = blockIdx.x;
  half_t* p = data + (size_t)row * H;
  float ss = 0.0f;
  for (int d = threadIdx.x; d < H; d += blockDim.x) { float x = (float)p[d]; ss += x * x; }
  ss = block_reduce<256>(ss);
  float inv = 1.0f / fmaxf(sqrtf(ss), 1e-12f);
  for (int d = threadIdx.x; d < H; d += blockDim.x) p[d] = (half_t)((float)p[d] * inv);
}

// saliency: softmax over token norms, one block per batch, 256 threads = tokens
__global__ void saliency_kernel(const float* z, int H, float* mu) {
  __shared__ float norms[NN];
  __shared__ float red[NN];
  int b = blockIdx.x, n = threadIdx.x;
  const float* row = z + ((size_t)b * NN + n) * H;
  float ss = 0.0f;
  for (int d = 0; d < H; ++d) { float x = row[d]; ss += x * x; }
  norms[n] = sqrtf(ss);
  __syncthreads();
  red[n] = norms[n]; __syncthreads();
  for (int s = NN >> 1; s > 0; s >>= 1) { if (n < s) red[n] = fmaxf(red[n], red[n + s]); __syncthreads(); }
  float mx = red[0]; __syncthreads();
  float e = expf(norms[n] - mx);
  red[n] = e; __syncthreads();
  for (int s = NN >> 1; s > 0; s >>= 1) { if (n < s) red[n] += red[n + s]; __syncthreads(); }
  mu[(size_t)b * NN + n] = e / red[0];
}

// in-batch cross entropy on 64x64 score matrix, fused
__global__ void scores_ce_kernel(const float* Q, const float* P, int dim, float w, float* acc) {
  __shared__ float srow[64];
  int i = blockIdx.x, j = threadIdx.x;
  const float* qi = Q + (size_t)i * dim;
  const float* pj = P + (size_t)j * dim;
  float dot = 0.0f;
  for (int d = 0; d < dim; ++d) dot += qi[d] * pj[d];
  srow[j] = dot * (1.0f / TEMPERATURE);
  __syncthreads();
  if (j == 0) {
    float mx = srow[0];
    for (int k = 1; k < 64; ++k) mx = fmaxf(mx, srow[k]);
    float s = 0.0f;
    for (int k = 0; k < 64; ++k) s += expf(srow[k] - mx);
    float lz = mx + logf(s);
    atomicAdd(acc + ACC_CONTRAST, w * (lz - srow[i]) * (1.0f / 64.0f));
  }
}

// squared pairwise distances of 128 x 128 matrix + upper-tri sum
__global__ void pairdist_kernel(const float* X, float* Dm, float* acc_sum) {
  int i = blockIdx.x, j = threadIdx.x;   // 128 threads
  const float* xi = X + (size_t)i * 128;
  const float* xj = X + (size_t)j * 128;
  float sqi = 0.0f, sqj = 0.0f, dot = 0.0f;
  for (int d = 0; d < 128; ++d) { float a = xi[d], b = xj[d]; sqi += a * a; sqj += b * b; dot += a * b; }
  float dv = sqi + sqj - 2.0f * dot;
  Dm[(size_t)i * 128 + j] = dv;
  float part = (j > i) ? dv : 0.0f;
  part = block_reduce<128>(part);
  if (j == 0) atomicAdd(acc_sum, part);
}

__global__ void rkd_dist_kernel(const float* Ds, const float* Dt, const float* acc_in, float* acc) {
  int i = blockIdx.x, j = threadIdx.x;
  const float cnt = 8128.0f;  // 128*127/2
  float ms = acc_in[ACC_DS_SUM] / cnt + 1e-8f;
  float mt = acc_in[ACC_DT_SUM] / cnt + 1e-8f;
  float part = 0.0f;
  if (j > i)
    part = huber_f(Ds[(size_t)i * 128 + j] / ms - Dt[(size_t)i * 128 + j] / mt);
  part = block_reduce<128>(part);
  if (j == 0) atomicAdd(acc + ACC_DIST, part);
}

// RKD angle: block j builds E_j = unit(x_i - x_j) in LDS (f16), then
// psi_j = E_j E_j^T via WMMA for both student and teacher; huber-mask-reduce.
__global__ void rkd_angle_kernel(const float* S, const float* T, float* acc) {
  __shared__ half_t Es[128 * 128];   // 32 KB
  __shared__ half_t Et[128 * 128];   // 32 KB
  int j = blockIdx.x, tid = threadIdx.x;
  if (tid < 128) {
    int i = tid; float ss = 0.0f;
    for (int d = 0; d < 128; ++d) { float df = S[(size_t)i * 128 + d] - S[(size_t)j * 128 + d]; ss += df * df; }
    float inv = 1.0f / (sqrtf(ss) + 1e-8f);
    for (int d = 0; d < 128; ++d)
      Es[i * 128 + d] = (half_t)((S[(size_t)i * 128 + d] - S[(size_t)j * 128 + d]) * inv);
  } else {
    int i = tid - 128; float ss = 0.0f;
    for (int d = 0; d < 128; ++d) { float df = T[(size_t)i * 128 + d] - T[(size_t)j * 128 + d]; ss += df * df; }
    float inv = 1.0f / (sqrtf(ss) + 1e-8f);
    for (int d = 0; d < 128; ++d)
      Et[i * 128 + d] = (half_t)((T[(size_t)i * 128 + d] - T[(size_t)j * 128 + d]) * inv);
  }
  __syncthreads();
  int lane = tid & 31, w = tid >> 5;
  float total = 0.0f;
  for (int t = w; t < 64; t += 8) {          // 8x8 tiles of 16x16, wave-uniform
    int tm = t >> 3, tn = t & 7;
    v8f cs = {}; v8f ct = {};
    for (int k0 = 0; k0 < 128; k0 += 32) {
      v16h as = load_a_frag(Es, 128, tm * 16, k0, lane);
      v16h bs = load_b_frag_t(Es, 128, k0, tn * 16, lane);
      cs = WMMA_F16(as, bs, cs);
      v16h at = load_a_frag(Et, 128, tm * 16, k0, lane);
      v16h bt = load_b_frag_t(Et, 128, k0, tn * 16, lane);
      ct = WMMA_F16(at, bt, ct);
    }
    int kcol  = tn * 16 + (lane & 15);
    int ibase = tm * 16 + 8 * (lane >> 4);
#pragma unroll
    for (int r = 0; r < 8; ++r) {
      int irow = ibase + r;
      if (irow != j && kcol != j && irow != kcol) total += huber_f(cs[r] - ct[r]);
    }
  }
  atomicAdd(acc + ACC_ANGLE, total);
}

// ---------------------------------------------------------------- staged GEMM
// C[b] = f(A[b] (rows_a x K) @ X[b](rows_x x K)^T), 128x128 macro-tile per
// workgroup, K staged 32 wide through double-buffered LDS with CDNA5 async
// copies (global_load_async_to_lds_b128 / s_wait_asynccnt), WMMA f16->f32.
// grid: (rows_x/128, rows_a/128, batches); 256 threads (8 waves).
template <int EPI>
__global__ void gemm_staged(const half_t* __restrict__ Abase,
                            const half_t* __restrict__ Xbase,
                            int K, int rows_a, int rows_x,
                            float* OutF, half_t* OutH, const float* bias,
                            const float* Cs, const float* mu, float* acc) {
  __shared__ half_t As[2][128 * 32];   // 8 KB per buffer
  __shared__ half_t Xs[2][128 * 32];
  int b  = blockIdx.z;
  int m0 = blockIdx.y * 128;
  int n0 = blockIdx.x * 128;
  const half_t* A = Abase + (size_t)b * rows_a * K;
  const half_t* X = Xbase + (size_t)b * rows_x * K;
  int tid = threadIdx.x, lane = tid & 31, w = tid >> 5;

  // staging: 8 KB tile = 256 threads * 32 B ; 64 B per row (32 halves)
  int o    = tid * 32;       // byte offset inside tile
  int srow = o >> 6;         // tile row
  int scol = o & 63;         // byte within row
  const char* gA0 = (const char*)(A + (size_t)(m0 + srow) * K) + scol;
  const char* gX0 = (const char*)(X + (size_t)(n0 + srow) * K) + scol;
  unsigned ldsA[2] = { lds_off_u32((const char*)&As[0][0] + o),
                       lds_off_u32((const char*)&As[1][0] + o) };
  unsigned ldsX[2] = { lds_off_u32((const char*)&Xs[0][0] + o),
                       lds_off_u32((const char*)&Xs[1][0] + o) };

  auto issue = [&](int kt, int buf) {
    const char* ga = gA0 + (size_t)kt * 64;   // 32 halves per k-step
    const char* gx = gX0 + (size_t)kt * 64;
    async_copy_b128(ldsA[buf],      ga);
    async_copy_b128(ldsA[buf] + 16, ga + 16);
    async_copy_b128(ldsX[buf],      gx);
    async_copy_b128(ldsX[buf] + 16, gx + 16);
    __builtin_prefetch(ga + 128, 0, 0);       // speculative, 2 stages ahead
    __builtin_prefetch(gx + 128, 0, 0);
  };

  int KT = K >> 5;
  v8f accv[8];
#pragma unroll
  for (int i = 0; i < 8; ++i) accv[i] = (v8f){};

  issue(0, 0);
  for (int kt = 0; kt < KT; ++kt) {
    int buf = kt & 1;
    if (kt + 1 < KT) {
      issue(kt + 1, buf ^ 1);
      wait_asynccnt<4>();       // newest 4 = next stage; current stage done
    } else {
      wait_asynccnt<0>();
    }
    __syncthreads();
    const half_t* Ab = &As[buf][0];
    const half_t* Xb = &Xs[buf][0];
    v16h af = load_a_frag(Ab, 32, w * 16, 0, lane);
#pragma unroll
    for (int nt = 0; nt < 8; ++nt) {
      v16h bf = load_b_frag_t(Xb, 32, 0, nt * 16, lane);
      accv[nt] = WMMA_F16(af, bf, accv[nt]);
    }
    __syncthreads();
  }

  // ---- epilogue ----
  int nlane = lane & 15;
  int mrow0 = m0 + w * 16 + 8 * (lane >> 4);
  float part = 0.0f;
#pragma unroll
  for (int nt = 0; nt < 8; ++nt) {
    int n = n0 + nt * 16 + nlane;
#pragma unroll
    for (int r = 0; r < 8; ++r) {
      int m = mrow0 + r;
      float cv = accv[nt][r];
      size_t idx = ((size_t)b * rows_a + m) * rows_x + n;
      if constexpr (EPI == EPI_1M_F32)   OutF[idx] = 1.0f - cv;
      if constexpr (EPI == EPI_1M_F16)   OutH[idx] = (half_t)(1.0f - cv);
      if constexpr (EPI == EPI_F16)      OutH[idx] = (half_t)cv;
      if constexpr (EPI == EPI_BIAS_F16) OutH[idx] = (half_t)(cv + bias[n]);
      if constexpr (EPI == EPI_STRUCT) {
        const float* mub = mu + (size_t)b * rows_a;
        float d = Cs[idx] - cv;
        part += d * d * mub[m] * mub[n];
      }
    }
  }
  if constexpr (EPI == EPI_STRUCT) atomicAdd(acc + ACC_STRUCT, part);
}

// ---------------------------------------------------------------- Sinkhorn
// One workgroup per batch: full 256x256 logK kept in 263 KB of LDS
// (CDNA5: up to 320 KB per workgroup). 20 log-domain iterations, then
// Gamma / row-normalized Gn (f16) and feat partial, all without re-reading HBM.
__global__ void sinkhorn_kernel(const float* Mbase, const float* mu, const float* nu,
                                half_t* Gn, float* acc) {
  extern __shared__ float lk[];                 // NN * (NN+1)
  __shared__ float u[NN], v[NN], lmu[NN], lnu[NN];
  const int rows = NN, stride = NN + 1;
  int b = blockIdx.x, tid = threadIdx.x;
  const float* Mb = Mbase + (size_t)b * rows * rows;
  for (int i = tid; i < rows * rows; i += blockDim.x) {
    int r = i >> 8, c = i & 255;
    lk[r * stride + c] = -Mb[i] * (1.0f / OT_EPS);
  }
  u[tid] = 0.0f; v[tid] = 0.0f;
  lmu[tid] = logf(mu[(size_t)b * rows + tid] + 1e-8f);
  lnu[tid] = logf(nu[(size_t)b * rows + tid] + 1e-8f);
  __syncthreads();
  for (int it = 0; it < OT_ITERS; ++it) {
    {   // u update: thread = n, LSE over m
      int n = tid;
      float mx = -3.0e38f;
      for (int m = 0; m < rows; ++m) mx = fmaxf(mx, lk[n * stride + m] + v[m]);
      float s = 0.0f;
      for (int m = 0; m < rows; ++m) s += expf(lk[n * stride + m] + v[m] - mx);
      u[n] = lmu[n] - (mx + logf(s));     // u not read in this phase
    }
    __syncthreads();
    {   // v update: thread = m, LSE over n
      int m = tid;
      float mx = -3.0e38f;
      for (int n = 0; n < rows; ++n) mx = fmaxf(mx, lk[n * stride + m] + u[n]);
      float s = 0.0f;
      for (int n = 0; n < rows; ++n) s += expf(lk[n * stride + m] + u[n] - mx);
      v[m] = lnu[m] - (mx + logf(s));
    }
    __syncthreads();
  }
  {   // Gamma row per thread; M = -eps*logK so no global rereads
    int n = tid;
    float rs = 0.0f, feat = 0.0f;
    for (int m = 0; m < rows; ++m) {
      float lkv = lk[n * stride + m];
      float g = expf(u[n] + lkv + v[m]);
      rs += g;
      feat += (-lkv * OT_EPS) * g;
      lk[n * stride + m] = g;             // own row: safe
    }
    float inv = 1.0f / (rs + 1e-8f);
    half_t* grow = Gn + ((size_t)b * rows + n) * rows;
    for (int m = 0; m < rows; ++m) grow[m] = (half_t)(lk[n * stride + m] * inv);
    atomicAdd(acc + ACC_FEAT, feat);
  }
}

__global__ void finalize_kernel(const float* acc, float* out) {
  float contrast = acc[ACC_CONTRAST];
  float dist  = acc[ACC_DIST]  / 8128.0f;        // C(128,2)
  float angle = acc[ACC_ANGLE] / 2048256.0f;     // 128*127*126
  float rkd   = 20.0f * (dist + 2.0f * angle);   // GLOBAL_RKD*2*(1*d + 2*a)
  // holo = 0.5 * sum_pair 0.5*(feat_mean + struct_mean) = (featsum+structsum)/(4*64)
  float holo  = (acc[ACC_FEAT] + acc[ACC_STRUCT]) * (1.0f / 256.0f);
  out[0] = contrast + rkd + holo;
}

// ---------------------------------------------------------------- launch

extern "C" void kernel_launch(void* const* d_in, const int* in_sizes, int n_in,
                              void* d_out, int out_size, void* d_ws, size_t ws_size,
                              hipStream_t stream) {
  (void)in_sizes; (void)n_in; (void)out_size; (void)ws_size;
  const float* s_q_reps   = (const float*)d_in[0];
  const float* s_p_reps   = (const float*)d_in[1];
  const float* t_q_reps   = (const float*)d_in[2];
  const float* t_p_reps   = (const float*)d_in[3];
  const float* s_q_states = (const float*)d_in[4];
  const float* s_p_states = (const float*)d_in[5];
  const float* t_q_states = (const float*)d_in[6];
  const float* t_p_states = (const float*)d_in[7];
  const float* proj_w     = (const float*)d_in[8];
  const float* proj_b     = (const float*)d_in[9];
  float* out = (float*)d_out;

  char* ws = (char*)d_ws;
  size_t off = 0;
  auto carve = [&](size_t bytes) -> char* {
    char* p = ws + off;
    off += (bytes + 255) & ~(size_t)255;
    return p;
  };
  float*  acc    = (float*)carve(32 * 4);
  float*  sbatch = (float*)carve(128 * 128 * 4);
  float*  tbatch = (float*)carve(128 * 128 * 4);
  float*  sqn768 = (float*)carve((size_t)64 * 768 * 4);
  float*  spn768 = (float*)carve((size_t)64 * 768 * 4);
  float*  dsmat  = (float*)carve(128 * 128 * 4);
  float*  dtmat  = (float*)carve(128 * 128 * 4);
  half_t* wt     = (half_t*)carve((size_t)HT * HS * 2);   // W^T, f16
  half_t* zsh    = (half_t*)carve((size_t)MROWS * HS * 2);
  half_t* zsn    = (half_t*)carve((size_t)MROWS * HS * 2);
  half_t* ztn    = (half_t*)carve((size_t)MROWS * HT * 2);
  half_t* Pn     = (half_t*)carve((size_t)MROWS * HT * 2);
  float*  muB    = (float*)carve((size_t)MROWS * 4);
  float*  nuB    = (float*)carve((size_t)MROWS * 4);
  float*  Cs     = (float*)carve((size_t)BB * NN * NN * 4);
  half_t* Cth    = (half_t*)carve((size_t)BB * NN * NN * 2);
  float*  Mmat   = (float*)carve((size_t)BB * NN * NN * 4);
  half_t* Gnh    = (half_t*)carve((size_t)BB * NN * NN * 2);
  half_t* T1h    = (half_t*)carve((size_t)BB * NN * NN * 2);

  zero_acc_kernel<<<1, 32, 0, stream>>>(acc);

  // ---- contrastive + RKD (dim 128 & 768) ----
  norm_slice_f32<<<64, 256, 0, stream>>>(s_q_reps, DREP, 128, sbatch, 128);
  norm_slice_f32<<<64, 256, 0, stream>>>(s_p_reps, DREP, 128, sbatch + 64 * 128, 128);
  norm_slice_f32<<<64, 256, 0, stream>>>(t_q_reps, DREP, 128, tbatch, 128);
  norm_slice_f32<<<64, 256, 0, stream>>>(t_p_reps, DREP, 128, tbatch + 64 * 128, 128);
  norm_slice_f32<<<64, 256, 0, stream>>>(s_q_reps, DREP, 768, sqn768, 768);
  norm_slice_f32<<<64, 256, 0, stream>>>(s_p_reps, DREP, 768, spn768, 768);

  scores_ce_kernel<<<64, 64, 0, stream>>>(sbatch, sbatch + 64 * 128, 128, 1.0f, acc);
  scores_ce_kernel<<<64, 64, 0, stream>>>(sqn768, spn768, 768, 2.0f, acc);

  pairdist_kernel<<<128, 128, 0, stream>>>(sbatch, dsmat, acc + ACC_DS_SUM);
  pairdist_kernel<<<128, 128, 0, stream>>>(tbatch, dtmat, acc + ACC_DT_SUM);
  rkd_dist_kernel<<<128, 128, 0, stream>>>(dsmat, dtmat, acc, acc);
  rkd_angle_kernel<<<128, 256, 0, stream>>>(sbatch, tbatch, acc);

  // ---- FGW (q then p), workspace reused ----
  transpose_w_kernel<<<dim3(HT / 32, HS / 32), 256, 0, stream>>>(proj_w, wt);

  const float* zs_list[2] = { s_q_states, s_p_states };
  const float* zt_list[2] = { t_q_states, t_p_states };
  for (int pair = 0; pair < 2; ++pair) {
    const float* zs = zs_list[pair];
    const float* zt = zt_list[pair];
    saliency_kernel<<<BB, 256, 0, stream>>>(zs, HS, muB);
    saliency_kernel<<<BB, 256, 0, stream>>>(zt, HT, nuB);
    rownorm_f32_to_f16<<<MROWS, 256, 0, stream>>>(zs, HS, zsn, zsh);
    rownorm_f32_to_f16<<<MROWS, 256, 0, stream>>>(zt, HT, ztn, (half_t*)nullptr);

    // P = z_s @ W + b   (as zsh (16384x768) @ Wt(1024x768)^T)
    gemm_staged<EPI_BIAS_F16><<<dim3(HT / 128, MROWS / 128, 1), 256, 0, stream>>>(
        zsh, wt, HS, MROWS, HT, nullptr, Pn, proj_b, nullptr, nullptr, nullptr);
    rownorm_inplace_f16<<<MROWS, 256, 0, stream>>>(Pn, HT);

    // C_s = 1 - zsn zsn^T (f32), C_t = 1 - ztn ztn^T (f16), M = 1 - Pn ztn^T (f32)
    gemm_staged<EPI_1M_F32><<<dim3(2, 2, BB), 256, 0, stream>>>(
        zsn, zsn, HS, NN, NN, Cs, nullptr, nullptr, nullptr, nullptr, nullptr);
    gemm_staged<EPI_1M_F16><<<dim3(2, 2, BB), 256, 0, stream>>>(
        ztn, ztn, HT, NN, NN, nullptr, Cth, nullptr, nullptr, nullptr, nullptr);
    gemm_staged<EPI_1M_F32><<<dim3(2, 2, BB), 256, 0, stream>>>(
        Pn, ztn, HT, NN, NN, Mmat, nullptr, nullptr, nullptr, nullptr, nullptr);

    sinkhorn_kernel<<<BB, 256, (size_t)NN * (NN + 1) * 4, stream>>>(Mmat, muB, nuB, Gnh, acc);

    // T1 = Gn @ C_t (C_t symmetric -> use B^T path); then fused struct loss
    gemm_staged<EPI_F16><<<dim3(2, 2, BB), 256, 0, stream>>>(
        Gnh, Cth, NN, NN, NN, nullptr, T1h, nullptr, nullptr, nullptr, nullptr);
    gemm_staged<EPI_STRUCT><<<dim3(2, 2, BB), 256, 0, stream>>>(
        T1h, Gnh, NN, NN, NN, nullptr, nullptr, nullptr, Cs, muB, acc);
  }

  finalize_kernel<<<1, 1, 0, stream>>>(acc, out);
}
